// MultiHeadSelfAttention_81501299409300
// MI455X (gfx1250) — matrix operand order repaired and checked by reading, not compiled, not run on hardware
//
#include <hip/hip_runtime.h>

typedef _Float16 f16;
typedef __attribute__((ext_vector_type(16))) _Float16 v16h;
typedef __attribute__((ext_vector_type(8)))  _Float16 v8h;
typedef __attribute__((ext_vector_type(8)))  float    v8f;

#define D_EMB 1024
#define NHEAD 16
#define HDIM  64
#define BATCH 2
#define SEQ   2048
#define M_TOT (BATCH*SEQ)   /* 4096 */
#define KDIM  1024
#define QK_SCALE 0.125f     /* 64^-0.5 */
#define LDSB  40            /* padded stride (halves) for 32-wide tiles */
#define LDSQ  72            /* padded stride (halves) for 64-wide tiles */

static __device__ __forceinline__ v8f vzero8() {
    v8f z;
#pragma unroll
    for (int i = 0; i < 8; ++i) z[i] = 0.0f;
    return z;
}

// Generic->LDS byte offset: for AMDGPU the generic address of an LDS object
// carries the wave-relative LDS offset in its low 32 bits.
static __device__ __forceinline__ unsigned lds_off(const void* p) {
    return (unsigned)(unsigned long long)(uintptr_t)p;
}

// CDNA5 async copy: global -> LDS, 16B per lane, tracked by ASYNCcnt.
static __device__ __forceinline__ void async_b128(unsigned lds, const void* g) {
    asm volatile("global_load_async_to_lds_b128 %0, %1, off"
                 :: "v"(lds), "v"((unsigned long long)(uintptr_t)g)
                 : "memory");
}
static __device__ __forceinline__ void wait_async0() {
    asm volatile("s_wait_asynccnt 0" ::: "memory");
}

// A-fragment (16x32 f16, M x K): lane m<16 holds row m, K{0..7}|{16..23};
// lanes 16-31 hold row m-16, K{8..15}|{24..31}.  row0 = first row of tile.
static __device__ __forceinline__ v16h ld_frag_a32(const f16* row0, int stride,
                                                   int lane, int koff) {
    const f16* p = row0 + (lane & 15) * stride + koff + ((lane >> 4) * 8);
    v8h lo = *(const v8h*)(p);
    v8h hi = *(const v8h*)(p + 16);
    v16h r;
#pragma unroll
    for (int i = 0; i < 8; ++i) { r[i] = lo[i]; r[i + 8] = hi[i]; }
    return r;
}

// B-fragment (32x16 f16, K x N) with B stored N-major (Bt[n][k]):
// lane n<16 holds col n, K{0..7}|{16..23}; lanes 16-31 hold col n-16, K{8..15}|{24..31}.
static __device__ __forceinline__ v16h ld_frag_b32(const f16* row0, int stride,
                                                   int lane, int koff) {
    const f16* p = row0 + (lane & 15) * stride + koff + ((lane >> 4) * 8);
    v8h lo = *(const v8h*)(p);
    v8h hi = *(const v8h*)(p + 16);
    v16h r;
#pragma unroll
    for (int i = 0; i < 8; ++i) { r[i] = lo[i]; r[i + 8] = hi[i]; }
    return r;
}

static __device__ __forceinline__ v8f wmma_f16(v16h a, v16h b, v8f c) {
    return __builtin_amdgcn_wmma_f32_16x16x32_f16(false, a, false, b,
                                                  (short)0, c, false, false);
}

// ---------------------------------------------------------------- converters
__global__ void cast_f32_to_f16(const float* __restrict__ src,
                                f16* __restrict__ dst, int n) {
    int i = blockIdx.x * 256 + threadIdx.x;
    if (i < n) dst[i] = (f16)src[i];
}

// dst[n*1024 + k] = (f16) src[k*N + n]   (K fixed at 1024)
__global__ void transpose_f32_to_f16(const float* __restrict__ src,
                                     f16* __restrict__ dst, int N) {
    int i = blockIdx.x * 256 + threadIdx.x;
    if (i < N * KDIM) {
        int n = i >> 10, kk = i & 1023;
        dst[i] = (f16)src[kk * N + n];
    }
}

// ---------------------------------------------------------------- QKV GEMM
// C[4096,3072] = A[4096,1024] x Bt[3072,1024]^T + bias, scattered into
// Q[B,H,L,Dh] (pre-scaled), K[B,H,L,Dh], Vt[B,H,Dh,L], all f16.
// Block tile 128(M) x 64(N), BK=32; per wave 32x64 (8 accumulators).
// Double-buffered LDS staged via async global->LDS copies; single loop body
// (unroll disabled) so accumulator tuples stay pinned across iterations.
__global__ __launch_bounds__(256)
void gemm_qkv(const f16* __restrict__ A, const f16* __restrict__ Bt,
              const float* __restrict__ bias,
              f16* __restrict__ qo, f16* __restrict__ ko, f16* __restrict__ vto) {
    __shared__ f16 as[2][128 * LDSB];
    __shared__ f16 bs[2][64 * LDSB];
    const int t = threadIdx.x, lane = t & 31, wave = t >> 5;
    const int n0 = blockIdx.x * 64;
    const int m0 = blockIdx.y * 128;

    v8f acc[2][4];
#pragma unroll
    for (int i = 0; i < 2; ++i)
#pragma unroll
        for (int j = 0; j < 4; ++j) acc[i][j] = vzero8();

    const int lrow = t >> 2, lseg = (t & 3) * 8;
    auto stage = [&](int buf, int k0) {
        async_b128(lds_off(&as[buf][lrow * LDSB + lseg]),
                   A + (size_t)(m0 + lrow) * KDIM + k0 + lseg);
        async_b128(lds_off(&as[buf][(lrow + 64) * LDSB + lseg]),
                   A + (size_t)(m0 + lrow + 64) * KDIM + k0 + lseg);
        async_b128(lds_off(&bs[buf][lrow * LDSB + lseg]),
                   Bt + (size_t)(n0 + lrow) * KDIM + k0 + lseg);
    };

    stage(0, 0);
    wait_async0();
    __syncthreads();

    const int NK = KDIM / 32;
    int cur = 0;
#pragma unroll 1
    for (int ks = 0; ks < NK; ++ks) {
        // Unconditional prefetch with wrap-around (tail prefetch is harmless)
        const int nxt = (ks + 1 == NK) ? 0 : ks + 1;
        stage(cur ^ 1, nxt * 32);

        const f16* ab = as[cur];
        const f16* bb = bs[cur];
        v16h af0 = ld_frag_a32(ab + (wave * 32) * LDSB, LDSB, lane, 0);
        v16h af1 = ld_frag_a32(ab + (wave * 32 + 16) * LDSB, LDSB, lane, 0);
#pragma unroll
        for (int nt = 0; nt < 4; ++nt) {
            v16h bf = ld_frag_b32(bb + nt * 16 * LDSB, LDSB, lane, 0);
            acc[0][nt] = wmma_f16(af0, bf, acc[0][nt]);
            acc[1][nt] = wmma_f16(af1, bf, acc[1][nt]);
        }
        wait_async0();
        __syncthreads();
        cur ^= 1;
    }

#pragma unroll
    for (int nt = 0; nt < 4; ++nt) {
        const int n = n0 + nt * 16 + (lane & 15);
        const int seg = n >> 10, c = n & 1023, h = c >> 6, d = c & 63;
        const float bv = bias[n];
#pragma unroll
        for (int mt = 0; mt < 2; ++mt) {
#pragma unroll
            for (int v = 0; v < 8; ++v) {
                const int m = m0 + wave * 32 + mt * 16 + v + ((lane >> 4) * 8);
                const int b = m >> 11, lq = m & 2047;
                const float val = acc[mt][nt][v] + bv;
                const size_t hl = (size_t)(b * NHEAD + h) * SEQ + lq;
                if (seg == 0)      qo[hl * HDIM + d] = (f16)(val * QK_SCALE);
                else if (seg == 1) ko[hl * HDIM + d] = (f16)val;
                else vto[((size_t)(b * NHEAD + h) * HDIM + d) * SEQ + lq] = (f16)val;
            }
        }
    }
}

// ---------------------------------------------------------------- attention
// One workgroup per (b,h, 64-query block); 4 waves, 16 q-rows each.
// Flash-style streaming over 32-key blocks with online softmax.
// K / Vt tiles double-buffered in LDS, staged by async copies so the next
// key block streams in while WMMA + softmax run on the current one.
__global__ __launch_bounds__(256)
void attn_kernel(const f16* __restrict__ q, const f16* __restrict__ k,
                 const f16* __restrict__ vt, f16* __restrict__ ctx) {
    __shared__ f16 qs[64 * LDSQ];
    __shared__ f16 ks2[2][32 * LDSQ];
    __shared__ f16 vts2[2][64 * LDSB];
    __shared__ f16 ps[4][16 * LDSB];

    const int t = threadIdx.x, lane = t & 31, wave = t >> 5;
    const int q0 = blockIdx.x * 64;
    const int bh = blockIdx.y;
    const f16* qp  = q  + (size_t)bh * SEQ * HDIM;
    const f16* kp  = k  + (size_t)bh * SEQ * HDIM;
    const f16* vtp = vt + (size_t)bh * HDIM * SEQ;

    auto stage_kv = [&](int buf, int kb) {
        {   // K block: 32 x 64 halves
            const int r = t >> 3, s = (t & 7) * 8;
            async_b128(lds_off(&ks2[buf][r * LDSQ + s]),
                       kp + (size_t)(kb * 32 + r) * HDIM + s);
        }
        {   // Vt block: 64 x 32 halves
            const int r = t >> 2, s = (t & 3) * 8;
            async_b128(lds_off(&vts2[buf][r * LDSB + s]),
                       vtp + (size_t)r * SEQ + kb * 32 + s);
        }
    };

#pragma unroll
    for (int i = 0; i < 2; ++i) {           // Q tile: 64 x 64 halves
        const int lin = t + i * 256;
        const int r = lin >> 3, s = (lin & 7) * 8;
        async_b128(lds_off(&qs[r * LDSQ + s]),
                   qp + (size_t)(q0 + r) * HDIM + s);
    }
    stage_kv(0, 0);
    wait_async0();
    __syncthreads();

    float m_run[8], l_run[8];
    v8f o[4];
#pragma unroll
    for (int v = 0; v < 8; ++v) { m_run[v] = -1e30f; l_run[v] = 0.0f; }
#pragma unroll
    for (int i = 0; i < 4; ++i) o[i] = vzero8();

    const int NB = SEQ / 32;
    int cur = 0;
#pragma unroll 1
    for (int kb = 0; kb < NB; ++kb) {
        const int nxt = (kb + 1 == NB) ? 0 : kb + 1;
        stage_kv(cur ^ 1, nxt);
        const f16* ksb = ks2[cur];
        const f16* vtb = vts2[cur];

        // S = Q Kt  (16 x 32 per wave, contraction over Dh=64 in 2 steps)
        v8f c0 = vzero8(), c1 = vzero8();
#pragma unroll
        for (int kk = 0; kk < 64; kk += 32) {
            v16h aq = ld_frag_a32(qs + wave * 16 * LDSQ, LDSQ, lane, kk);
            v16h b0 = ld_frag_b32(ksb, LDSQ, lane, kk);
            v16h b1 = ld_frag_b32(ksb + 16 * LDSQ, LDSQ, lane, kk);
            c0 = wmma_f16(aq, b0, c0);
            c1 = wmma_f16(aq, b1, c1);
        }

        // online softmax; C-layout row lives across 16 lanes of a half-wave
        f16* pw = ps[wave];
#pragma unroll
        for (int v = 0; v < 8; ++v) {
            const float s0 = c0[v], s1 = c1[v];
            float mx = fmaxf(s0, s1);
#pragma unroll
            for (int off = 1; off < 16; off <<= 1)
                mx = fmaxf(mx, __shfl_xor(mx, off, 32));
            const float mn = fmaxf(m_run[v], mx);
            const float alpha = __expf(m_run[v] - mn);
            const float p0 = __expf(s0 - mn), p1 = __expf(s1 - mn);
            float ls = p0 + p1;
#pragma unroll
            for (int off = 1; off < 16; off <<= 1)
                ls += __shfl_xor(ls, off, 32);
            l_run[v] = l_run[v] * alpha + ls;
            m_run[v] = mn;
#pragma unroll
            for (int nt = 0; nt < 4; ++nt) o[nt][v] *= alpha;
            const int ml = v + ((lane >> 4) * 8);
            pw[ml * LDSB + (lane & 15)]      = (f16)p0;
            pw[ml * LDSB + (lane & 15) + 16] = (f16)p1;
        }

        // O += P V   (A-frag from per-wave LDS, B-frag from Vt tile)
        v16h ap = ld_frag_a32(pw, LDSB, lane, 0);
#pragma unroll
        for (int nt = 0; nt < 4; ++nt) {
            v16h bv = ld_frag_b32(vtb + nt * 16 * LDSB, LDSB, lane, 0);
            o[nt] = wmma_f16(ap, bv, o[nt]);
        }
        wait_async0();
        __syncthreads();
        cur ^= 1;
    }

    const int b = bh >> 4, h = bh & 15;
#pragma unroll
    for (int v = 0; v < 8; ++v) {
        const float inv = 1.0f / l_run[v];
        const int m = q0 + wave * 16 + v + ((lane >> 4) * 8);
        const size_t rowoff = (size_t)(b * SEQ + m) * D_EMB + h * HDIM;
#pragma unroll
        for (int nt = 0; nt < 4; ++nt)
            ctx[rowoff + nt * 16 + (lane & 15)] = (f16)(o[nt][v] * inv);
    }
}

// ---------------------------------------------------------------- out proj
// Same tiling as gemm_qkv; plain f32 + bias epilogue.
__global__ __launch_bounds__(256)
void gemm_out(const f16* __restrict__ A, const f16* __restrict__ Bt,
              const float* __restrict__ bias, float* __restrict__ out) {
    __shared__ f16 as[2][128 * LDSB];
    __shared__ f16 bs[2][64 * LDSB];
    const int t = threadIdx.x, lane = t & 31, wave = t >> 5;
    const int n0 = blockIdx.x * 64;
    const int m0 = blockIdx.y * 128;

    v8f acc[2][4];
#pragma unroll
    for (int i = 0; i < 2; ++i)
#pragma unroll
        for (int j = 0; j < 4; ++j) acc[i][j] = vzero8();

    const int lrow = t >> 2, lseg = (t & 3) * 8;
    auto stage = [&](int buf, int k0) {
        async_b128(lds_off(&as[buf][lrow * LDSB + lseg]),
                   A + (size_t)(m0 + lrow) * KDIM + k0 + lseg);
        async_b128(lds_off(&as[buf][(lrow + 64) * LDSB + lseg]),
                   A + (size_t)(m0 + lrow + 64) * KDIM + k0 + lseg);
        async_b128(lds_off(&bs[buf][lrow * LDSB + lseg]),
                   Bt + (size_t)(n0 + lrow) * KDIM + k0 + lseg);
    };

    stage(0, 0);
    wait_async0();
    __syncthreads();

    const int NK = KDIM / 32;
    int cur = 0;
#pragma unroll 1
    for (int ks = 0; ks < NK; ++ks) {
        const int nxt = (ks + 1 == NK) ? 0 : ks + 1;
        stage(cur ^ 1, nxt * 32);

        const f16* ab = as[cur];
        const f16* bb = bs[cur];
        v16h af0 = ld_frag_a32(ab + (wave * 32) * LDSB, LDSB, lane, 0);
        v16h af1 = ld_frag_a32(ab + (wave * 32 + 16) * LDSB, LDSB, lane, 0);
#pragma unroll
        for (int nt = 0; nt < 4; ++nt) {
            v16h bf = ld_frag_b32(bb + nt * 16 * LDSB, LDSB, lane, 0);
            acc[0][nt] = wmma_f16(af0, bf, acc[0][nt]);
            acc[1][nt] = wmma_f16(af1, bf, acc[1][nt]);
        }
        wait_async0();
        __syncthreads();
        cur ^= 1;
    }

#pragma unroll
    for (int nt = 0; nt < 4; ++nt) {
        const int n = n0 + nt * 16 + (lane & 15);
        const float bv = bias[n];
#pragma unroll
        for (int mt = 0; mt < 2; ++mt) {
#pragma unroll
            for (int v = 0; v < 8; ++v) {
                const int m = m0 + wave * 32 + mt * 16 + v + ((lane >> 4) * 8);
                out[(size_t)m * D_EMB + n] = acc[mt][nt][v] + bv;
            }
        }
    }
}

// ---------------------------------------------------------------- launch
extern "C" void kernel_launch(void* const* d_in, const int* in_sizes, int n_in,
                              void* d_out, int out_size, void* d_ws, size_t ws_size,
                              hipStream_t stream) {
    const float* x      = (const float*)d_in[0];
    const float* w_qkv  = (const float*)d_in[1];
    const float* b_qkv  = (const float*)d_in[2];
    const float* w_proj = (const float*)d_in[3];
    const float* b_proj = (const float*)d_in[4];
    float* out = (float*)d_out;

    // Workspace carve-out (~48 MB of f16 staging), 256B aligned chunks.
    char* ws = (char*)d_ws;
    size_t off = 0;
    auto carve = [&](size_t bytes) {
        char* p = ws + off;
        off += (bytes + 255) & ~(size_t)255;
        return p;
    };
    f16* xh     = (f16*)carve((size_t)M_TOT * D_EMB * sizeof(f16));
    f16* wqkvT  = (f16*)carve((size_t)3 * D_EMB * D_EMB * sizeof(f16));
    f16* wprojT = (f16*)carve((size_t)D_EMB * D_EMB * sizeof(f16));
    f16* qh     = (f16*)carve((size_t)M_TOT * D_EMB * sizeof(f16));
    f16* kh     = (f16*)carve((size_t)M_TOT * D_EMB * sizeof(f16));
    f16* vth    = (f16*)carve((size_t)M_TOT * D_EMB * sizeof(f16));
    f16* ctx    = (f16*)carve((size_t)M_TOT * D_EMB * sizeof(f16));
    (void)ws_size; (void)in_sizes; (void)n_in; (void)out_size;

    cast_f32_to_f16<<<(M_TOT * D_EMB + 255) / 256, 256, 0, stream>>>(
        x, xh, M_TOT * D_EMB);
    transpose_f32_to_f16<<<(3 * D_EMB * KDIM + 255) / 256, 256, 0, stream>>>(
        w_qkv, wqkvT, 3 * D_EMB);
    transpose_f32_to_f16<<<(D_EMB * KDIM + 255) / 256, 256, 0, stream>>>(
        w_proj, wprojT, D_EMB);

    gemm_qkv<<<dim3(3 * D_EMB / 64, M_TOT / 128), 256, 0, stream>>>(
        xh, wqkvT, b_qkv, qh, kh, vth);

    attn_kernel<<<dim3(SEQ / 64, BATCH * NHEAD), 256, 0, stream>>>(
        qh, kh, vth, ctx);

    gemm_out<<<dim3(D_EMB / 64, M_TOT / 128), 256, 0, stream>>>(
        ctx, wprojT, b_proj, out);
}
